// AttentionLayer_29119878266976
// MI455X (gfx1250) — matrix operand order repaired and checked by reading, not compiled, not run on hardware
//
#include <hip/hip_runtime.h>

// ---------------------------------------------------------------------------
// CDNA5 (gfx1250) wave32 WMMA attention layer:
//   q = (X Wq^T + bq) * scale*log2e   (f32 -> f16, double-buffered WMMA GEMM)
//   k =  X Wk^T + bk                  (f32 -> f16)
//   vT = (X Wv^T + bv) transposed per head to (B,H,E,S)  (f32 -> f16)
//   flash attention, causal, online softmax in exp2 domain (WMMA QK^T + PV),
//     K/V^T slabs staged by the Tensor Data Mover (tensor_load_to_lds) with
//     hardware LDS row padding; TENSORcnt-synchronized ping-pong.
//   out = ctx Wo^T + bo               (f16 -> f32)
// ---------------------------------------------------------------------------

typedef __attribute__((ext_vector_type(4)))  _Float16 v4h;
typedef __attribute__((ext_vector_type(8)))  _Float16 v8h;
typedef __attribute__((ext_vector_type(16))) _Float16 v16h;
typedef __attribute__((ext_vector_type(8)))  float    v8f;
typedef __attribute__((ext_vector_type(4)))  unsigned int v4u;
typedef __attribute__((ext_vector_type(8)))  int          v8i;
typedef __attribute__((ext_vector_type(4)))  int          v4i;

union V16H { v16h v; v8h h[2]; };

constexpr int AB = 4, AL = 2048, AD = 1024, AH = 16, AE = 64;

#if __has_builtin(__builtin_amdgcn_tensor_load_to_lds) && \
    __has_builtin(__builtin_amdgcn_s_wait_tensorcnt)
#define USE_TDM 1
#else
#define USE_TDM 0
#endif

// A-matrix fragment (16x32 f16): lane group g=lane>>4 takes K runs
// [g*8 .. g*8+7] and [g*8+16 .. g*8+23] from the row owned by lane&15.
__device__ __forceinline__ v16h load_fragA(const _Float16* rowp, int lane) {
  V16H u;
  const _Float16* p = rowp + ((lane >> 4) << 3);
  u.h[0] = *(const v8h*)(p);
  u.h[1] = *(const v8h*)(p + 16);
  return u.v;
}

// B-matrix fragment (32x16 f16): lane owns column n = lane&15 (B^T row n),
// lane group g takes K runs [g*8 .. g*8+7] and [g*8+16 .. g*8+23].
__device__ __forceinline__ v16h load_fragB(const _Float16* rowp, int lane) {
  V16H u;
  const _Float16* p = rowp + ((lane >> 4) << 3);
  u.h[0] = *(const v8h*)(p);
  u.h[1] = *(const v8h*)(p + 16);
  return u.v;
}

#if USE_TDM
// TDM: DMA a 64x64 f16 tile (row stride `stride_elems` halfs) from global to
// LDS at byte offset `lds_off`, inserting 4 DWORDs of padding every 32 DWORDs
// (=> 72-half LDS rows matching SKP). D# layout per CDNA5 ISA 8.3/8.4.
// This toolchain's builtin is the 6-arg form:
//   (uint32x4 g0, int32x8 g1, int32x4 g2, int32x4 g3, int32x8, i32 cpol)
__device__ __forceinline__ void tdm_load_64x64(unsigned lds_off,
                                               const void* gptr,
                                               unsigned stride_elems) {
  unsigned long long ga = (unsigned long long)gptr;
  v4u g0 = { 1u,                                  // count=1 (valid, user D#)
             lds_off,                              // lds_addr (bytes)
             (unsigned)ga,                         // global_addr[31:0]
             (unsigned)(ga >> 32) | 0x80000000u }; // ga[56:32] | type=2<<30
  v8i g1 = { (int)((1u << 16)    // data_size = 2 bytes
                 | (1u << 20)    // pad_enable
                 | (4u << 22)    // pad_interval: every 32 DWORDs
                 | (3u << 25)),  // pad_amount: 4 DWORDs
             (int)(64u << 16),   // tensor_dim0 = 64 (lo16 in hi half)
             (int)(64u << 16),   // tensor_dim0 hi=0 | tensor_dim1 = 64
             (int)(64u << 16),   // tensor_dim1 hi=0 | tile_dim0 = 64
             64,                 // tile_dim1 = 64 | tile_dim2 = 0
             (int)stride_elems,  // tensor_dim0_stride[31:0]
             0, 0 };             // stride hi / dim1_stride = 0
  v4i gz4 = { 0, 0, 0, 0 };
  v8i gz8 = { 0, 0, 0, 0, 0, 0, 0, 0 };
  __builtin_amdgcn_tensor_load_to_lds(g0, g1, gz4, gz4, gz8, 0);
}
#endif

// ---------------------------------------------------------------------------
// GEMM: C[M,N] = (A[M,K] @ W[N,K]^T + bias[N]) * oscale
// 256 threads = 8 waves (2M x 4N); wave tile 64x32; ping-pong LDS double
// buffer with register-staged prefetch -> one barrier per 32-K slab.
// (VGPR staging kept here because tiles need f32->f16 conversion.)
// ---------------------------------------------------------------------------
constexpr int GBM = 128, GBN = 128, GBK = 32, LDT = 40; // LDT: padded halfs/row

template <bool AF16, bool CF16, bool VT>
__global__ __launch_bounds__(256)
void gemm_xwt(const void* __restrict__ Ap, const float* __restrict__ W,
              const float* __restrict__ bias, void* __restrict__ Cp,
              int M, int N, int K, float oscale)
{
  __shared__ _Float16 As[2][GBM * LDT];   // 2 x 10 KB
  __shared__ _Float16 Bs[2][GBN * LDT];   // 2 x 10 KB

  const int tid  = threadIdx.x;
  const int lane = tid & 31;
  const int wave = tid >> 5;
  const int wm   = wave >> 2;   // 0..1
  const int wn   = wave & 3;    // 0..3
  const int bm0  = blockIdx.y * GBM;
  const int bn0  = blockIdx.x * GBN;

  v8f acc[4][2] = {};

  float4 ga[4]; v8h ga16[2]; float4 gw[4];

  auto loadG = [&](int k0) {
    if constexpr (AF16) {
      const _Float16* A16 = (const _Float16*)Ap;
      #pragma unroll
      for (int i = 0; i < 2; ++i) {
        int idx = tid + i * 256, row = idx >> 2, c8 = idx & 3;
        ga16[i] = *(const v8h*)(A16 + (size_t)(bm0 + row) * K + k0 + c8 * 8);
      }
    } else {
      const float* A32 = (const float*)Ap;
      #pragma unroll
      for (int i = 0; i < 4; ++i) {
        int idx = tid + i * 256, row = idx >> 3, c4 = idx & 7;
        ga[i] = *(const float4*)(A32 + (size_t)(bm0 + row) * K + k0 + c4 * 4);
      }
    }
    #pragma unroll
    for (int i = 0; i < 4; ++i) {
      int idx = tid + i * 256, row = idx >> 3, c4 = idx & 7;
      gw[i] = *(const float4*)(W + (size_t)(bn0 + row) * K + k0 + c4 * 4);
    }
  };

  auto storeL = [&](int buf) {
    if constexpr (AF16) {
      #pragma unroll
      for (int i = 0; i < 2; ++i) {
        int idx = tid + i * 256, row = idx >> 2, c8 = idx & 3;
        *(v8h*)&As[buf][row * LDT + c8 * 8] = ga16[i];
      }
    } else {
      #pragma unroll
      for (int i = 0; i < 4; ++i) {
        int idx = tid + i * 256, row = idx >> 3, c4 = idx & 7;
        v4h hc = { (_Float16)ga[i].x, (_Float16)ga[i].y,
                   (_Float16)ga[i].z, (_Float16)ga[i].w };
        *(v4h*)&As[buf][row * LDT + c4 * 4] = hc;
      }
    }
    #pragma unroll
    for (int i = 0; i < 4; ++i) {
      int idx = tid + i * 256, row = idx >> 3, c4 = idx & 7;
      v4h hc = { (_Float16)gw[i].x, (_Float16)gw[i].y,
                 (_Float16)gw[i].z, (_Float16)gw[i].w };
      *(v4h*)&Bs[buf][row * LDT + c4 * 4] = hc;
    }
  };

  loadG(0);
  storeL(0);
  __syncthreads();

  const int nk = K / GBK;
  for (int kb = 0; kb < nk; ++kb) {
    const int cur = kb & 1;
    if (kb + 1 < nk) loadG((kb + 1) * GBK);   // global in flight over compute

    v16h a[4], bfr[2];
    #pragma unroll
    for (int tm = 0; tm < 4; ++tm)
      a[tm] = load_fragA(&As[cur][(wm * 64 + tm * 16 + (lane & 15)) * LDT], lane);
    #pragma unroll
    for (int tn = 0; tn < 2; ++tn)
      bfr[tn] = load_fragB(&Bs[cur][(wn * 32 + tn * 16 + (lane & 15)) * LDT], lane);
    #pragma unroll
    for (int tm = 0; tm < 4; ++tm)
      #pragma unroll
      for (int tn = 0; tn < 2; ++tn)
        acc[tm][tn] = __builtin_amdgcn_wmma_f32_16x16x32_f16(
            false, a[tm], false, bfr[tn], (short)0, acc[tm][tn], false, false);

    if (kb + 1 < nk) storeL(cur ^ 1);
    __syncthreads();
  }

  // ---- epilogue: bias, scale, store (optionally per-head-transposed V) ----
  #pragma unroll
  for (int tm = 0; tm < 4; ++tm) {
    #pragma unroll
    for (int tn = 0; tn < 2; ++tn) {
      int n = bn0 + wn * 32 + tn * 16 + (lane & 15);
      float bnv = bias[n];
      #pragma unroll
      for (int r = 0; r < 8; ++r) {
        int m = bm0 + wm * 64 + tm * 16 + r + ((lane >> 4) << 3);
        float val = (acc[tm][tn][r] + bnv) * oscale;
        if constexpr (VT) {
          int bb = m / AL, ss = m % AL, hh = n / AE, ee = n % AE;
          ((_Float16*)Cp)[((size_t)((bb * AH + hh) * AE + ee)) * AL + ss] =
              (_Float16)val;
        } else if constexpr (CF16) {
          ((_Float16*)Cp)[(size_t)m * N + n] = (_Float16)val;
        } else {
          ((float*)Cp)[(size_t)m * N + n] = val;
        }
      }
    }
  }
}

// ---------------------------------------------------------------------------
// Flash attention (causal). Q pre-scaled by scale*log2e -> softmax in exp2
// domain. V pre-transposed (B,H,E,S). K/V^T slabs staged by the TDM into a
// single flat LDS arena (compile-time byte offsets for the D#), ping-pong
// buffered with one barrier per 64-key slab; wave 0 issues the DMA and
// synchronizes on TENSORcnt. Mask only on the two diagonal slabs.
// ---------------------------------------------------------------------------
constexpr int SKP   = 72;            // padded halfs per LDS row
constexpr int SLABH = 64 * SKP;      // halfs per K or V^T slab buffer

__global__ __launch_bounds__(256)
void flash_attn(const _Float16* __restrict__ q, const _Float16* __restrict__ k,
                const _Float16* __restrict__ vT, _Float16* __restrict__ ctx)
{
  // flat arena: [ sK0 | sK1 | sV0 | sV1 | sP(8 waves x 16 rows) ]
  __shared__ _Float16 smem[4 * SLABH + 8 * 16 * SKP];   // 55296 B
  _Float16* SK = smem;                  // sK[buf] = SK + buf*SLABH
  _Float16* SV = smem + 2 * SLABH;      // sV[buf] = SV + buf*SLABH
  _Float16* SP = smem + 4 * SLABH;      // sP[wave] = SP + wave*16*SKP

  const int tid  = threadIdx.x;
  const int lane = tid & 31;
  const int wave = tid >> 5;
  const int lrow = lane & 15;
  const int hig  = lane >> 4;

  const int qb = blockIdx.x;               // query block (128 rows)
  const int bh = blockIdx.y;
  const int b  = bh >> 4, h = bh & 15;

  const _Float16* qh = q  + (size_t)b * AL * AD + h * AE;
  const _Float16* kh = k  + (size_t)b * AL * AD + h * AE;
  const _Float16* vh = vT + (size_t)(b * AH + h) * AE * AL;

  const int qi0 = qb * 128 + wave * 16;

  // Persistent (pre-scaled) Q fragments
  v16h aq[2];
  {
    const _Float16* qp = qh + (size_t)(qi0 + lrow) * AD;
    #pragma unroll
    for (int kt = 0; kt < 2; ++kt) aq[kt] = load_fragA(qp + kt * 32, lane);
  }

  float mrow[8], lsum[8];
  v8f o[4] = {};
  #pragma unroll
  for (int r = 0; r < 8; ++r) { mrow[r] = -1e30f; lsum[r] = 0.0f; }

#if !USE_TDM
  v8h gk[2], gv[2];                        // fallback register staging
#endif

  auto issue_slab = [&](int j, int buf) {
#if USE_TDM
    if (wave == 0) {
      tdm_load_64x64((unsigned)(buf * SLABH) * 2u,
                     kh + (size_t)(j * 64) * AD, AD);
      tdm_load_64x64((unsigned)(2 * SLABH + buf * SLABH) * 2u,
                     vh + (size_t)(j * 64), AL);
    }
#else
    #pragma unroll
    for (int i = 0; i < 2; ++i) {
      int idx = tid + i * 256, row = idx >> 3, c8 = idx & 7;
      gk[i] = *(const v8h*)(kh + (size_t)(j * 64 + row) * AD + c8 * 8);
      gv[i] = *(const v8h*)(vh + (size_t)row * AL + j * 64 + c8 * 8);
    }
    (void)buf;
#endif
  };
  auto commit_slab = [&](int buf) {
#if USE_TDM
    if (wave == 0) __builtin_amdgcn_s_wait_tensorcnt(0);
    (void)buf;
#else
    #pragma unroll
    for (int i = 0; i < 2; ++i) {
      int idx = tid + i * 256, row = idx >> 3, c8 = idx & 7;
      *(v8h*)&SK[buf * SLABH + row * SKP + c8 * 8] = gk[i];
      *(v8h*)&SV[buf * SLABH + row * SKP + c8 * 8] = gv[i];
    }
#endif
  };

  const int nblk = 2 * qb + 2;             // causal: keys up to qb*128+127
  issue_slab(0, 0);
  commit_slab(0);
  __syncthreads();

  for (int j = 0; j < nblk; ++j) {
    const int cur = j & 1;
    if (j + 1 < nblk) issue_slab(j + 1, cur ^ 1);  // DMA/global in flight

    // ---- S = Q K^T (pre-scaled to exp2 domain), 8 WMMAs ----
    v8f s[4] = {};
    #pragma unroll
    for (int tn = 0; tn < 4; ++tn)
      #pragma unroll
      for (int kt = 0; kt < 2; ++kt) {
        v16h bb = load_fragB(&SK[cur * SLABH + (tn * 16 + lrow) * SKP + kt * 32],
                             lane);
        s[tn] = __builtin_amdgcn_wmma_f32_16x16x32_f16(
            false, aq[kt], false, bb, (short)0, s[tn], false, false);
      }

    // ---- causal mask: only the two diagonal slabs (uniform branch) ----
    if (j >= 2 * qb) {
      #pragma unroll
      for (int tn = 0; tn < 4; ++tn) {
        int kj = j * 64 + tn * 16 + lrow;
        #pragma unroll
        for (int r = 0; r < 8; ++r) {
          int qi = qi0 + r + (hig << 3);
          if (kj > qi) s[tn][r] = -1e30f;
        }
      }
    }

    // ---- online softmax in exp2 domain ----
    float rm[8], ps[8];
    #pragma unroll
    for (int r = 0; r < 8; ++r) {
      float x = fmaxf(fmaxf(s[0][r], s[1][r]), fmaxf(s[2][r], s[3][r]));
      x = fmaxf(x, __shfl_xor(x, 1, 32));
      x = fmaxf(x, __shfl_xor(x, 2, 32));
      x = fmaxf(x, __shfl_xor(x, 4, 32));
      x = fmaxf(x, __shfl_xor(x, 8, 32));
      rm[r] = fmaxf(mrow[r], x);
      ps[r] = 0.0f;
    }
    #pragma unroll
    for (int tn = 0; tn < 4; ++tn)
      #pragma unroll
      for (int r = 0; r < 8; ++r) {
        float p = exp2f(s[tn][r] - rm[r]);
        s[tn][r] = p;
        ps[r] += p;
      }
    #pragma unroll
    for (int r = 0; r < 8; ++r) {
      float x = ps[r];
      x += __shfl_xor(x, 1, 32);
      x += __shfl_xor(x, 2, 32);
      x += __shfl_xor(x, 4, 32);
      x += __shfl_xor(x, 8, 32);
      float alpha = exp2f(mrow[r] - rm[r]);
      lsum[r] = lsum[r] * alpha + x;
      mrow[r] = rm[r];
      #pragma unroll
      for (int t = 0; t < 4; ++t) o[t][r] *= alpha;
    }

    // ---- P: C-layout -> f16 A-layout via wave-private LDS ----
    _Float16* pw = &SP[wave * 16 * SKP];
    #pragma unroll
    for (int tn = 0; tn < 4; ++tn)
      #pragma unroll
      for (int r = 0; r < 8; ++r)
        pw[(r + (hig << 3)) * SKP + tn * 16 + lrow] = (_Float16)s[tn][r];

    v16h ap[2];
    #pragma unroll
    for (int kt = 0; kt < 2; ++kt)
      ap[kt] = load_fragA(pw + lrow * SKP + kt * 32, lane);

    // ---- O += P V (8 WMMAs; B^T rows are V^T rows in sV) ----
    #pragma unroll
    for (int te = 0; te < 4; ++te)
      #pragma unroll
      for (int kt = 0; kt < 2; ++kt) {
        v16h bb = load_fragB(&SV[cur * SLABH + (te * 16 + lrow) * SKP + kt * 32],
                             lane);
        o[te] = __builtin_amdgcn_wmma_f32_16x16x32_f16(
            false, ap[kt], false, bb, (short)0, o[te], false, false);
      }

    if (j + 1 < nblk) commit_slab(cur ^ 1);
    __syncthreads();
  }

  // ---- normalize + write context (B, L, H*E) as f16 ----
  #pragma unroll
  for (int te = 0; te < 4; ++te) {
    int e = te * 16 + lrow;
    #pragma unroll
    for (int r = 0; r < 8; ++r) {
      int qi = qi0 + r + (hig << 3);
      float valr = o[te][r] / lsum[r];
      ctx[((size_t)b * AL + qi) * AD + h * AE + e] = (_Float16)valr;
    }
  }
}

// ---------------------------------------------------------------------------
// Launcher
// ---------------------------------------------------------------------------
extern "C" void kernel_launch(void* const* d_in, const int* in_sizes, int n_in,
                              void* d_out, int out_size, void* d_ws, size_t ws_size,
                              hipStream_t stream)
{
  (void)in_sizes; (void)n_in; (void)out_size; (void)ws_size;

  const float* queries = (const float*)d_in[0];
  const float* keys    = (const float*)d_in[1];
  const float* values  = (const float*)d_in[2];
  // d_in[3] = attn_mask (causal, known statically) -- unused
  const float* Wq = (const float*)d_in[4];
  const float* bq = (const float*)d_in[5];
  const float* Wk = (const float*)d_in[6];
  const float* bk = (const float*)d_in[7];
  const float* Wv = (const float*)d_in[8];
  const float* bv = (const float*)d_in[9];
  const float* Wo = (const float*)d_in[10];
  const float* bo = (const float*)d_in[11];

  const int M = AB * AL;   // 8192
  const int N = AD;        // 1024
  const int K = AD;        // 1024

  _Float16* q16  = (_Float16*)d_ws;
  _Float16* k16  = q16 + (size_t)M * N;
  _Float16* vT16 = k16 + (size_t)M * N;
  _Float16* c16  = vT16 + (size_t)M * N;   // total 64 MB of workspace

  // scale*log2e folded into q so softmax runs natively in exp2 domain
  const float qscale = 0.125f * 1.44269504088896f;

  dim3 gg(N / GBN, M / GBM), blk(256);
  gemm_xwt<false, true,  false><<<gg, blk, 0, stream>>>(queries, Wq, bq, q16,  M, N, K, qscale);
  gemm_xwt<false, true,  false><<<gg, blk, 0, stream>>>(keys,    Wk, bk, k16,  M, N, K, 1.0f);
  gemm_xwt<false, true,  true ><<<gg, blk, 0, stream>>>(values,  Wv, bv, vT16, M, N, K, 1.0f);

  flash_attn<<<dim3(AL / 128, AB * AH), blk, 0, stream>>>(q16, k16, vT16, c16);

  gemm_xwt<true,  false, false><<<gg, blk, 0, stream>>>(c16, Wo, bo, (float*)d_out, M, N, K, 1.0f);
}
